// VectorQuantizer_58119497449850
// MI455X (gfx1250) — compile-verified, matrix operand order checked by reading, not run on hardware
//
#include <hip/hip_runtime.h>

// ---------------------------------------------------------------------------
// VQ (cosine codebook search) for MI455X / gfx1250, wave32 + WMMA bf16.
//   x:        [32,1024,256] f32   (d_in[0])
//   embedding:[8192,256]    f32   (d_in[1])
//   out: z_out (8388608 f32) | loss (1 f32) | idx (32768, stored as float)
//
// Phase 2 v3: two register-resident 16x256 A tiles per wave (32 rows), B
// fragment shared by two WMMAs, and B double-buffered so LDS loads for step
// k+1 are in flight while step k's WMMA pair executes (no full dscnt drain).
// ---------------------------------------------------------------------------

#define M_ROWS   32768      // 32*1024
#define DIMS     256
#define NCODES   8192
#define CHUNK    64         // codes staged in LDS per iteration
#define NCHUNK   (NCODES / CHUNK)
#define LDS_ROW  264        // bf16 elems per LDS row (528B: rows step 4 banks)
#define ROWS_PER_WAVE 32
#define ROWS_PER_BLOCK (8 * ROWS_PER_WAVE)

typedef __bf16 v16bf __attribute__((ext_vector_type(16)));
typedef __bf16 v8bf  __attribute__((ext_vector_type(8)));
typedef float  v8f   __attribute__((ext_vector_type(8)));

__device__ __forceinline__ float block_reduce_sum_256(float v, float* red) {
    const int lane = threadIdx.x & 31;
    const int wave = threadIdx.x >> 5;
    #pragma unroll
    for (int off = 16; off; off >>= 1) v += __shfl_xor(v, off, 32);
    if (lane == 0) red[wave] = v;
    __syncthreads();
    float tot = 0.f;
    #pragma unroll
    for (int i = 0; i < 8; ++i) tot += red[i];
    __syncthreads();            // allow caller to reuse `red`
    return tot;
}

// Build a 16-element bf16 WMMA fragment from two 16-byte chunks at the ISA's
// 16-bit operand layout offsets (K chunk of 32, per-lane half selection).
__device__ __forceinline__ v16bf load_frag(const __bf16* vec, int kc, int half) {
    v8bf lo = *(const v8bf*)(vec + kc * 32 + half * 8);
    v8bf hi = *(const v8bf*)(vec + kc * 32 + 16 + half * 8);
    return __builtin_shufflevector(lo, hi,
            0,1,2,3,4,5,6,7,8,9,10,11,12,13,14,15);
}

// ---- Phase 1a: normalize codebook -> f32 + bf16; zero the loss slot -------
__global__ void vq_norm_cb(const float* __restrict__ emb,
                           float* __restrict__ cbn,
                           __bf16* __restrict__ cbb,
                           float* __restrict__ loss_slot) {
    __shared__ float red[8];
    const int row = blockIdx.x;
    const int t   = threadIdx.x;
    if (row == 0 && t == 0) *loss_slot = 0.f;   // before phase 3 (stream order)
    const float v = emb[row * DIMS + t];
    const float tot = block_reduce_sum_256(v * v, red);
    const float inv = 1.f / fmaxf(sqrtf(tot), 1e-12f);
    const float o = v * inv;
    cbn[row * DIMS + t] = o;
    cbb[row * DIMS + t] = (__bf16)o;
}

// ---- Phase 1b: normalize x -> bf16 ----------------------------------------
__global__ void vq_norm_x(const float* __restrict__ x,
                          __bf16* __restrict__ xnb) {
    __shared__ float red[8];
    const int row = blockIdx.x;
    const int t   = threadIdx.x;
    const float v = x[row * DIMS + t];
    const float tot = block_reduce_sum_256(v * v, red);
    const float inv = 1.f / fmaxf(sqrtf(tot), 1e-12f);
    xnb[row * DIMS + t] = (__bf16)(v * inv);
}

// ---- Phase 2: WMMA argmax sweep -------------------------------------------
__global__ __launch_bounds__(256)
void vq_argmax(const __bf16* __restrict__ xnb,
               const __bf16* __restrict__ cbb,
               int* __restrict__ idx_ws) {
    __shared__ __bf16 lds[CHUNK * LDS_ROW];      // 33,792 B

    const int t    = threadIdx.x;
    const int wave = t >> 5;
    const int lane = t & 31;
    const int half = lane >> 4;
    const int l16  = lane & 15;
    const int rowBase = (blockIdx.x * 8 + wave) * ROWS_PER_WAVE;

    // Load two 16x256 A tiles into registers (rows rowBase.. and rowBase+16..).
    v16bf A0[8], A1[8];
    {
        const __bf16* xr0 = xnb + (size_t)(rowBase + l16) * DIMS;
        const __bf16* xr1 = xr0 + 16 * DIMS;
        #pragma unroll
        for (int kc = 0; kc < 8; ++kc) {
            A0[kc] = load_frag(xr0, kc, half);
            A1[kc] = load_frag(xr1, kc, half);
        }
    }

    float best0[8], best1[8];
    int   bidx0[8], bidx1[8];
    #pragma unroll
    for (int v = 0; v < 8; ++v) {
        best0[v] = -3.0e38f; bidx0[v] = 0;
        best1[v] = -3.0e38f; bidx1[v] = 0;
    }

    for (int chunk = 0; chunk < NCHUNK; ++chunk) {
        __syncthreads();
        // Cooperative stage: CHUNK rows x 512B -> LDS (padded rows).
        #pragma unroll
        for (int i = 0; i < 8; ++i) {
            const int seg = t + i * 256;          // 0..2047 16-byte segments
            const int r = seg >> 5;
            const int c = seg & 31;
            v8bf d = *(const v8bf*)(cbb + (size_t)(chunk * CHUNK + r) * DIMS + c * 8);
            *(v8bf*)(&lds[r * LDS_ROW + c * 8]) = d;
        }
        if (chunk + 1 < NCHUNK)                   // global_prefetch next chunk
            __builtin_prefetch(cbb + (size_t)(chunk + 1) * CHUNK * DIMS + t * 64, 0, 0);
        __syncthreads();

        #pragma unroll
        for (int tile = 0; tile < CHUNK / 16; ++tile) {
            const __bf16* brow = &lds[(tile * 16 + l16) * LDS_ROW];
            v8f c0 = {};
            v8f c1 = {};
            // B ping-pong: fragment for step kc+1 is loaded before the step
            // kc WMMA pair consumes Bcur, keeping 2 DS loads in flight.
            v16bf Bcur = load_frag(brow, 0, half);
            #pragma unroll
            for (int kc = 0; kc < 8; ++kc) {
                v16bf Bnext;
                if (kc < 7) {
                    Bnext = load_frag(brow, kc + 1, half);
                } else if (tile < CHUNK / 16 - 1) {
                    // Prime the next tile's first fragment to keep the
                    // LDS pipe busy across the accumulator readout.
                    Bnext = load_frag(brow + 16 * LDS_ROW, 0, half);
                } else {
                    Bnext = Bcur;
                }
                c0 = __builtin_amdgcn_wmma_f32_16x16x32_bf16(
                        false, A0[kc], false, Bcur, (short)0, c0, false, false);
                c1 = __builtin_amdgcn_wmma_f32_16x16x32_bf16(
                        false, A1[kc], false, Bcur, (short)0, c1, false, false);
                Bcur = Bnext;
            }
            const int n = chunk * CHUNK + tile * 16 + l16;
            #pragma unroll
            for (int v = 0; v < 8; ++v) {
                if (c0[v] > best0[v]) { best0[v] = c0[v]; bidx0[v] = n; }
                if (c1[v] > best1[v]) { best1[v] = c1[v]; bidx1[v] = n; }
            }
        }
    }

    // Cross-lane argmax within each 16-lane group (ties -> lowest index,
    // matching argmin's first-occurrence semantics).
    #pragma unroll
    for (int off = 8; off >= 1; off >>= 1) {
        #pragma unroll
        for (int v = 0; v < 8; ++v) {
            float ob0 = __shfl_xor(best0[v], off, 32);
            int   oi0 = __shfl_xor(bidx0[v], off, 32);
            if (ob0 > best0[v] || (ob0 == best0[v] && oi0 < bidx0[v])) {
                best0[v] = ob0; bidx0[v] = oi0;
            }
            float ob1 = __shfl_xor(best1[v], off, 32);
            int   oi1 = __shfl_xor(bidx1[v], off, 32);
            if (ob1 > best1[v] || (ob1 == best1[v] && oi1 < bidx1[v])) {
                best1[v] = ob1; bidx1[v] = oi1;
            }
        }
    }
    if (l16 == 0) {
        #pragma unroll
        for (int v = 0; v < 8; ++v) {
            idx_ws[rowBase + half * 8 + v]      = bidx0[v];
            idx_ws[rowBase + 16 + half * 8 + v] = bidx1[v];
        }
    }
}

// ---- Phase 3: gather z_out, exact f32 loss, idx ---------------------------
__global__ void vq_output(const float* __restrict__ x,
                          const float* __restrict__ cbn,
                          const int* __restrict__ idx_ws,
                          float* __restrict__ z_out,
                          float* __restrict__ loss,
                          float* __restrict__ idx_out) {
    __shared__ float red[8];
    const int row = blockIdx.x;
    const int t   = threadIdx.x;

    const float xv  = x[row * DIMS + t];
    const float tot = block_reduce_sum_256(xv * xv, red);
    const float inv = 1.f / fmaxf(sqrtf(tot), 1e-12f);
    const float xn  = xv * inv;

    const int id  = idx_ws[row];
    const float q = cbn[(size_t)id * DIMS + t];
    z_out[row * DIMS + t] = q;

    const float d  = q - xn;
    const float sq = block_reduce_sum_256(d * d, red);
    if (t == 0) {
        atomicAdd(loss, sq * (1.25f / (float)(M_ROWS * DIMS)));
        idx_out[row] = (float)id;
    }
}

// ---------------------------------------------------------------------------
extern "C" void kernel_launch(void* const* d_in, const int* in_sizes, int n_in,
                              void* d_out, int out_size, void* d_ws, size_t ws_size,
                              hipStream_t stream) {
    const float* x   = (const float*)d_in[0];   // 32*1024*256
    const float* emb = (const float*)d_in[1];   // 8192*256

    char* ws = (char*)d_ws;
    float*  cbn    = (float*)ws;                               // 8 MB
    ws += (size_t)NCODES * DIMS * sizeof(float);
    __bf16* cbb    = (__bf16*)ws;                              // 4 MB
    ws += (size_t)NCODES * DIMS * sizeof(__bf16);
    __bf16* xnb    = (__bf16*)ws;                              // 16 MB
    ws += (size_t)M_ROWS * DIMS * sizeof(__bf16);
    int*    idx_ws = (int*)ws;                                 // 128 KB

    float* z_out   = (float*)d_out;
    float* loss    = z_out + (size_t)M_ROWS * DIMS;
    float* idx_out = loss + 1;

    vq_norm_cb<<<NCODES, 256, 0, stream>>>(emb, cbn, cbb, loss);
    vq_norm_x <<<M_ROWS, 256, 0, stream>>>(x, xnb);
    vq_argmax <<<M_ROWS / ROWS_PER_BLOCK, 256, 0, stream>>>(xnb, cbb, idx_ws);
    vq_output <<<M_ROWS, 256, 0, stream>>>(x, cbn, idx_ws, z_out, loss, idx_out);
}
